// RelationModule_50414326120627
// MI455X (gfx1250) — matrix operand order repaired and checked by reading, not compiled, not run on hardware
//
#include <hip/hip_runtime.h>

typedef __attribute__((ext_vector_type(16))) _Float16 v16h;
typedef __attribute__((ext_vector_type(8)))  float    v8f;

#define H_  4
#define N_  37
#define NP  48   // rows padded to 3x16
#define NS  40   // score column stride (>=37)
#define DF_ 64
#define DK_ 8
#define DV_ 16

__global__ __launch_bounds__(128) void relation_kernel(
    const float* __restrict__ f_a, const float* __restrict__ Wk,
    const float* __restrict__ Wq,  const float* __restrict__ Wv,
    const float* __restrict__ w_fix, float* __restrict__ out)
{
    __shared__ float    s_fa[NP * DF_];       // 12288 B  staged f_a (padded)
    __shared__ _Float16 s_k[H_][NP * DK_];    //  3072 B
    __shared__ _Float16 s_q[H_][NP * DK_];    //  3072 B
    __shared__ float    s_v[H_][NP * DV_];    // 12288 B
    __shared__ float    s_s[H_][NP * NS];     // 30720 B  scores
    __shared__ float    s_rs[H_][NP];         //   768 B  row sums
    // total ~62 KB LDS

    const int b    = blockIdx.x;
    const int tid  = threadIdx.x;
    const int lane = tid & 31;
    const int h    = tid >> 5;          // wave id == head

    // ---- stage f_a[b] (37x64) into LDS, zero-pad rows 37..47 ----
    const float* fa_g = f_a + (size_t)b * (N_ * DF_);
    for (int idx = tid; idx < NP * DF_; idx += 128)
        s_fa[idx] = (idx < N_ * DF_) ? fa_g[idx] : 0.0f;
    __syncthreads();

    const int row16 = lane & 15;
    const int hi    = (lane >> 4) & 1;  // 0: lanes 0-15, 1: lanes 16-31
    const int kb8   = hi * 8;           // A-fragment K base
    const int kb16  = hi * 16;          // unused; B fragments use kb8

    // ---- B fragments for Wk/Wq/Wv (independent of row tile) ----
    // B layout (16-bit, 32x16): lane holds column (lane&15), K = (j>>3)*16 + kb8 + (j&7)
    v16h bk0 = {}, bk1 = {}, bq0 = {}, bq1 = {}, bv0 = {}, bv1 = {};
    {
        const int col = row16;
        const float* wk = Wk + h * DF_ * DK_;
        const float* wq = Wq + h * DF_ * DK_;
        const float* wv = Wv + h * DF_ * DV_;
#pragma unroll
        for (int j = 0; j < 16; ++j) {
            const int f0 = (j >> 3) * 16 + kb8 + (j & 7);         // k-step 0
            const int f1 = 32 + (j >> 3) * 16 + kb8 + (j & 7);    // k-step 1
            bk0[j] = (col < DK_) ? (_Float16)wk[f0 * DK_ + col] : (_Float16)0.0f;
            bk1[j] = (col < DK_) ? (_Float16)wk[f1 * DK_ + col] : (_Float16)0.0f;
            bq0[j] = (col < DK_) ? (_Float16)wq[f0 * DK_ + col] : (_Float16)0.0f;
            bq1[j] = (col < DK_) ? (_Float16)wq[f1 * DK_ + col] : (_Float16)0.0f;
            bv0[j] = (_Float16)wv[f0 * DV_ + col];
            bv1[j] = (_Float16)wv[f1 * DV_ + col];
        }
    }

    // ---- projections k,q,v = f_a @ W via v_wmma_f32_16x16x32_f16 ----
    // A layout (16-bit, 16x32): lane holds row (lane&15); K = (j>>3)*16 + kb8 + (j&7)
#pragma unroll
    for (int rt = 0; rt < 3; ++rt) {
        const int arow = rt * 16 + row16;
        v16h a0 = {}, a1 = {};
#pragma unroll
        for (int j = 0; j < 16; ++j) {
            const int K = (j >> 3) * 16 + kb8 + (j & 7);
            a0[j] = (_Float16)s_fa[arow * DF_ + K];
            a1[j] = (_Float16)s_fa[arow * DF_ + 32 + K];
        }
        v8f ck = {}, cq = {}, cv = {};
        ck = __builtin_amdgcn_wmma_f32_16x16x32_f16(false, a0, false, bk0, (short)0, ck, false, false);
        ck = __builtin_amdgcn_wmma_f32_16x16x32_f16(false, a1, false, bk1, (short)0, ck, false, false);
        cq = __builtin_amdgcn_wmma_f32_16x16x32_f16(false, a0, false, bq0, (short)0, cq, false, false);
        cq = __builtin_amdgcn_wmma_f32_16x16x32_f16(false, a1, false, bq1, (short)0, cq, false, false);
        cv = __builtin_amdgcn_wmma_f32_16x16x32_f16(false, a0, false, bv0, (short)0, cv, false, false);
        cv = __builtin_amdgcn_wmma_f32_16x16x32_f16(false, a1, false, bv1, (short)0, cv, false, false);
        // D layout: lane holds column (lane&15); row M = r + 8*hi
#pragma unroll
        for (int r = 0; r < 8; ++r) {
            const int m = rt * 16 + r + hi * 8;
            const int n = row16;
            if (n < DK_) {
                s_k[h][m * DK_ + n] = (_Float16)ck[r];
                s_q[h][m * DK_ + n] = (_Float16)cq[r];
            }
            s_v[h][m * DV_ + n] = cv[r];
        }
    }

    // ---- scores = (k q^T)/sqrt(DK), K=8 zero-padded to 32, via WMMA ----
    const float scl = 0.35355339059327373f;  // 1/sqrt(8)
#pragma unroll
    for (int rt = 0; rt < 3; ++rt) {
        v16h ak = {};
#pragma unroll
        for (int j = 0; j < 16; ++j) {
            const int K = (j >> 3) * 16 + kb8 + (j & 7);
            ak[j] = (K < DK_) ? s_k[h][(rt * 16 + row16) * DK_ + K] : (_Float16)0.0f;
        }
#pragma unroll
        for (int ct = 0; ct < 3; ++ct) {
            v16h bq = {};
#pragma unroll
            for (int j = 0; j < 16; ++j) {
                const int K = (j >> 3) * 16 + kb8 + (j & 7);
                bq[j] = (K < DK_) ? s_q[h][(ct * 16 + row16) * DK_ + K] : (_Float16)0.0f;
            }
            v8f cs = {};
            cs = __builtin_amdgcn_wmma_f32_16x16x32_f16(false, ak, false, bq, (short)0, cs, false, false);
#pragma unroll
            for (int r = 0; r < 8; ++r) {
                const int m = rt * 16 + r + hi * 8;
                const int n = ct * 16 + row16;
                if (n < NS) s_s[h][m * NS + n] = cs[r] * scl;
            }
        }
    }

    // ---- masked softmax normalizer + w_fix row sums (per-lane rows) ----
    // rowsum_m = (1 - softmax[m,m]) + sum_{n!=m} w_fix[h,m,n]
    const float* wfh = w_fix + h * N_ * N_;
    for (int m = lane; m < N_; m += 32) {
        float mx = 0.0f;  // diagonal masked to 0 participates in the max
        for (int n = 0; n < N_; ++n) {
            const float s = (n == m) ? 0.0f : s_s[h][m * NS + n];
            mx = fmaxf(mx, s);
        }
        float Z = 0.0f;
        for (int n = 0; n < N_; ++n) {
            const float s = (n == m) ? 0.0f : s_s[h][m * NS + n];
            Z += __expf(s - mx);
        }
        const float pdiag = __expf(-mx) / Z;
        float fs = 0.0f;
        for (int n = 0; n < N_; ++n)
            fs += (n == m) ? 0.0f : wfh[m * N_ + n];
        s_rs[h][m] = 1.0f - pdiag + fs;
    }

    // ---- out[b,m,h*16+j] = v[h,m,j] * rowsum[m] + f_a[b,m,h*16+j] ----
    float* og = out + (size_t)b * (N_ * DF_);
    for (int idx = lane; idx < N_ * DV_; idx += 32) {
        const int m = idx >> 4;
        const int j = idx & 15;
        const float val = s_v[h][m * DV_ + j] * s_rs[h][m] + s_fa[m * DF_ + h * DV_ + j];
        og[m * DF_ + h * DV_ + j] = val;
    }
}

extern "C" void kernel_launch(void* const* d_in, const int* in_sizes, int n_in,
                              void* d_out, int out_size, void* d_ws, size_t ws_size,
                              hipStream_t stream) {
    (void)n_in; (void)out_size; (void)d_ws; (void)ws_size;
    const float* f_a   = (const float*)d_in[0];
    const float* Wk    = (const float*)d_in[1];
    const float* Wq    = (const float*)d_in[2];
    const float* Wv    = (const float*)d_in[3];
    const float* w_fix = (const float*)d_in[4];
    float* out = (float*)d_out;
    const int Bn = in_sizes[0] / (N_ * DF_);
    relation_kernel<<<dim3(Bn), dim3(128), 0, stream>>>(f_a, Wk, Wq, Wv, w_fix, out);
}
